// tfdiff_ModRec_69827578298855
// MI455X (gfx1250) — matrix operand order repaired and checked by reading, not compiled, not run on hardware
//
#include <hip/hip_runtime.h>
#include <math.h>

#define DEVFN __device__ __forceinline__

constexpr int cB = 2, cS = 2048, cH = 256, cNH = 8, cDH = 32, cL = 4;
constexpr int cCOND = 64, cEMB = 256, cMAXSTEP = 1000, cH4 = 1024;
constexpr int cM = cB * cS;          // 4096 rows
constexpr int LDP = 40;              // LDS row pitch in u16 (80B -> 16B aligned)

typedef __attribute__((ext_vector_type(16))) __bf16 v16bf;
typedef __attribute__((ext_vector_type(8)))  float  v8f;

union Frag { unsigned int u[8]; unsigned short h[16]; v16bf v; };

DEVFN unsigned short f2bf(float f) {
  union { float f; unsigned int u; } v; v.f = f;
  unsigned int r = v.u + 0x7FFFu + ((v.u >> 16) & 1u);
  return (unsigned short)(r >> 16);
}
DEVFN float siluf(float x) { return x / (1.f + __expf(-x)); }
DEVFN float nan0(float x) { return (x != x) ? 0.f : x; }
DEVFN float clampf(float x, float lo, float hi) { return fminf(fmaxf(x, lo), hi); }

DEVFN v8f bwmma(const Frag& a, const Frag& b, v8f c) {
  return __builtin_amdgcn_wmma_f32_16x16x32_bf16(false, a.v, false, b.v, (short)0, c,
                                                 false, false);
}

// CDNA5 async global->LDS copy, 16B per lane; tracked by ASYNCcnt.
// Low 32 bits of a flat shared-aperture pointer are the LDS byte offset (ISA 10.2).
DEVFN void async_ld16(void* lds, const void* gmem) {
  unsigned l = (unsigned)(size_t)lds;
  unsigned long long g = (unsigned long long)(size_t)gmem;
  asm volatile("global_load_async_to_lds_b128 %0, %1, off" ::"v"(l), "v"(g) : "memory");
}
DEVFN void async_wait0() { asm volatile("s_wait_asynccnt 0x0" ::: "memory"); }

DEVFN float blockSum(float v, float* sbuf) {
  int lane = threadIdx.x & 31, w = threadIdx.x >> 5;
#pragma unroll
  for (int o = 16; o > 0; o >>= 1) v += __shfl_xor(v, o, 32);
  if (lane == 0) sbuf[w] = v;
  __syncthreads();
  if (w == 0) {
    float r = (lane < 8) ? sbuf[lane] : 0.f;
#pragma unroll
    for (int o = 4; o > 0; o >>= 1) r += __shfl_xor(r, o, 32);
    if (lane == 0) sbuf[0] = r;
  }
  __syncthreads();
  float res = sbuf[0];
  __syncthreads();
  return res;
}

// ---------------------------------------------------------------- utilities
__global__ void k_f32_to_bf16(const float* __restrict__ src,
                              unsigned short* __restrict__ dst, int n) {
  int i = blockIdx.x * 256 + threadIdx.x;
  if (i < n) dst[i] = f2bf(src[i]);
}

// h0 = cmul(clin(x, p_lin), pe)   -> f32 planes [M,H]
__global__ __launch_bounds__(256) void k_embed(
    const float* __restrict__ x,
    const float* __restrict__ pWr, const float* __restrict__ pWi,
    const float* __restrict__ pbr, const float* __restrict__ pbi,
    float* __restrict__ hr, float* __restrict__ hi) {
  int bs = blockIdx.x, hh = threadIdx.x;
  int s = bs % cS;
  float xr = x[bs * 2 + 0], xi = x[bs * 2 + 1];
  float wr = pWr[hh], wi = pWi[hh];
  float ar = wr * xr - wi * xi + pbr[hh];
  float ai = wi * xr + wr * xi + pbi[hh];
  float ang = (float)s * __expf(-logf((float)cS) * (float)hh / (float)cH);
  float co = cosf(ang), sn = sinf(ang);
  hr[(size_t)bs * cH + hh] = ar * co - ai * sn;
  hi[(size_t)bs * cH + hh] = ar * sn + ai * co;
}

// timestep table lookup: te[b][e] = (cos,sin)(t[b]*exp(-log(1000)*e/EMB))
__global__ void k_te(const int* __restrict__ t,
                     float* __restrict__ ter, float* __restrict__ tei) {
  int b = blockIdx.y, e = threadIdx.x;
  float ang = (float)t[b] * __expf(-logf((float)cMAXSTEP) * (float)e / (float)cEMB);
  ter[b * cEMB + e] = cosf(ang);
  tei[b * cEMB + e] = sinf(ang);
}

// small complex GEMV: y[b][N] = clin(act(x[b]), W(N,K)) (+accum)
__global__ void k_clin_vec(
    const float* __restrict__ xr, const float* __restrict__ xi, int xstride,
    const float* __restrict__ Wr, const float* __restrict__ Wi,
    const float* __restrict__ br, const float* __restrict__ bi,
    float* __restrict__ yr, float* __restrict__ yi, int ystride,
    int N, int K, int silu_in, int interleaved, int accum) {
  int b = blockIdx.y;
  int n = blockIdx.x * blockDim.x + threadIdx.x;
  if (n >= N) return;
  float ar = 0.f, ai = 0.f;
  for (int k = 0; k < K; k++) {
    float vr, vi;
    if (interleaved) {
      vr = xr[((size_t)b * xstride + k) * 2 + 0];
      vi = xr[((size_t)b * xstride + k) * 2 + 1];
    } else {
      vr = xr[(size_t)b * xstride + k];
      vi = xi[(size_t)b * xstride + k];
    }
    if (silu_in) { vr = siluf(vr); vi = siluf(vi); }
    float wr = Wr[(size_t)n * K + k], wi = Wi[(size_t)n * K + k];
    ar = fmaf(vr, wr, ar); ar = fmaf(-vi, wi, ar);
    ai = fmaf(vi, wr, ai); ai = fmaf(vr, wi, ai);
  }
  ar += br[n]; ai += bi[n];
  if (accum) { yr[(size_t)b * ystride + n] += ar; yi[(size_t)b * ystride + n] += ai; }
  else       { yr[(size_t)b * ystride + n]  = ar; yi[(size_t)b * ystride + n]  = ai; }
}

// LayerNorm over H (per component) + modulate -> bf16 mx planes
__global__ __launch_bounds__(256) void k_ln_mod(
    const float* __restrict__ hr, const float* __restrict__ hi,
    const float* __restrict__ modR, const float* __restrict__ modI,
    int cShift, int cScale,
    unsigned short* __restrict__ mxr, unsigned short* __restrict__ mxi) {
  __shared__ float sbuf[8];
  int bs = blockIdx.x, hh = threadIdx.x, b = bs / cS;
  float vr = hr[(size_t)bs * cH + hh], vi = hi[(size_t)bs * cH + hh];
  float mr = blockSum(vr, sbuf) * (1.f / cH);
  float mi = blockSum(vi, sbuf) * (1.f / cH);
  float qr = blockSum(vr * vr, sbuf) * (1.f / cH);
  float qi = blockSum(vi * vi, sbuf) * (1.f / cH);
  float nr = (vr - mr) * rsqrtf(qr - mr * mr + 1e-6f);
  float ni = (vi - mi) * rsqrtf(qi - mi * mi + 1e-6f);
  nr = nan0(nr); ni = nan0(ni);
  const float* mR = modR + (size_t)b * 6 * cH;
  const float* mI = modI + (size_t)b * 6 * cH;
  float shr = mR[cShift * cH + hh], shi = mI[cShift * cH + hh];
  float scr = clampf(mR[cScale * cH + hh], -5.f, 5.f);
  float sci = clampf(mI[cScale * cH + hh], -5.f, 5.f);
  mxr[(size_t)bs * cH + hh] = f2bf(nan0(nr * (1.f + scr) + shr));
  mxi[(size_t)bs * cH + hh] = f2bf(nan0(ni * (1.f + sci) + shi));
}

// --------------------------------------------------------- complex WMMA GEMM
// C = A[M,K](complex bf16) x W(N,K)^T(complex bf16) + bias
// WG tile 128x64 (8 waves, 4x2), 32x32 complex tile per wave -> 16 wmma/K-step.
// OUTMODE 0: store bf16 (optional SiLU)    OUTMODE 1: h += gate * C (f32)
template <int OUTMODE, int SILUO>
__global__ __launch_bounds__(256) void k_cgemm(
    const unsigned short* __restrict__ Ar, const unsigned short* __restrict__ Ai,
    const unsigned short* __restrict__ Wr, const unsigned short* __restrict__ Wi,
    const float* __restrict__ br, const float* __restrict__ bi,
    unsigned short* __restrict__ Or, unsigned short* __restrict__ Oi,
    float* __restrict__ Hr, float* __restrict__ Hi,
    const float* __restrict__ gR, const float* __restrict__ gI, int gstride,
    int N, int K) {
  __shared__ unsigned short sAr[128][LDP], sAi[128][LDP];
  __shared__ unsigned short sWr[64][LDP], sWi[64][LDP];
  int tid = threadIdx.x, lane = tid & 31, w = tid >> 5;
  int wm = w & 3, wn = w >> 2;
  int m0 = blockIdx.y * 128, n0 = blockIdx.x * 64;
  int half = lane >> 4, lm = lane & 15;
  v8f aR[2][2] = {};
  v8f aI[2][2] = {};
  for (int k0 = 0; k0 < K; k0 += 32) {
    // async-stage A (128x32, 2 planes) and W (64x32, 2 planes) directly to LDS
#pragma unroll
    for (int it = 0; it < 2; ++it) {
      int slot = tid + 256 * it, r = slot >> 2, c8 = (slot & 3) * 8;
      async_ld16(&sAr[r][c8], Ar + (size_t)(m0 + r) * K + k0 + c8);
      async_ld16(&sAi[r][c8], Ai + (size_t)(m0 + r) * K + k0 + c8);
    }
    {
      int r = tid >> 2, c8 = (tid & 3) * 8;
      async_ld16(&sWr[r][c8], Wr + (size_t)(n0 + r) * K + k0 + c8);
      async_ld16(&sWi[r][c8], Wi + (size_t)(n0 + r) * K + k0 + c8);
    }
    async_wait0();
    __syncthreads();
    Frag fAr[2], fAi[2], fAiN[2], fWrF[2], fWiF[2];
#pragma unroll
    for (int ms = 0; ms < 2; ms++) {
      const unsigned int* ra = (const unsigned int*)&sAr[32 * wm + 16 * ms + lm][0];
      const unsigned int* rb = (const unsigned int*)&sAi[32 * wm + 16 * ms + lm][0];
#pragma unroll
      for (int vg = 0; vg < 8; vg++) {
        int p = vg + 8 * half;
        fAr[ms].u[vg] = ra[p];
        fAi[ms].u[vg] = rb[p];
        fAiN[ms].u[vg] = rb[p] ^ 0x80008000u;
      }
    }
#pragma unroll
    for (int ns = 0; ns < 2; ns++) {
      const unsigned int* rw = (const unsigned int*)&sWr[32 * wn + 16 * ns + lm][0];
      const unsigned int* rwi = (const unsigned int*)&sWi[32 * wn + 16 * ns + lm][0];
#pragma unroll
      for (int j = 0; j < 8; j++) {
        fWrF[ns].u[j] = rw[j + 8 * half];
        fWiF[ns].u[j] = rwi[j + 8 * half];
      }
    }
#pragma unroll
    for (int ms = 0; ms < 2; ms++)
#pragma unroll
      for (int ns = 0; ns < 2; ns++) {
        aR[ms][ns] = bwmma(fAr[ms], fWrF[ns], aR[ms][ns]);
        aR[ms][ns] = bwmma(fAiN[ms], fWiF[ns], aR[ms][ns]);
        aI[ms][ns] = bwmma(fAi[ms], fWrF[ns], aI[ms][ns]);
        aI[ms][ns] = bwmma(fAr[ms], fWiF[ns], aI[ms][ns]);
      }
    __syncthreads();
  }
#pragma unroll
  for (int ms = 0; ms < 2; ms++)
#pragma unroll
    for (int ns = 0; ns < 2; ns++) {
      int n = n0 + 32 * wn + 16 * ns + lm;
      float biasR = br[n], biasI = bi[n];
#pragma unroll
      for (int r = 0; r < 8; r++) {
        int m = m0 + 32 * wm + 16 * ms + r + 8 * half;
        float yr = aR[ms][ns][r] + biasR, yi = aI[ms][ns][r] + biasI;
        if (OUTMODE == 0) {
          if (SILUO) { yr = siluf(yr); yi = siluf(yi); }
          Or[(size_t)m * N + n] = f2bf(yr);
          Oi[(size_t)m * N + n] = f2bf(yi);
        } else {
          int b = m / cS;
          Hr[(size_t)m * N + n] += gR[(size_t)b * gstride + n] * yr;
          Hi[(size_t)m * N + n] += gI[(size_t)b * gstride + n] * yi;
        }
      }
    }
}

// ------------------------------------------------------ flash attention WMMA
__global__ __launch_bounds__(256) void k_attn(
    const unsigned short* __restrict__ qr, const unsigned short* __restrict__ qi,
    const unsigned short* __restrict__ kr, const unsigned short* __restrict__ ki,
    const unsigned short* __restrict__ vr, const unsigned short* __restrict__ vi,
    unsigned short* __restrict__ aor, unsigned short* __restrict__ aoi) {
  __shared__ unsigned short sK[2][32][LDP];
  __shared__ unsigned short sV[2][32][LDP];
  __shared__ unsigned short sP[8][16][LDP];
  int tid = threadIdx.x, lane = tid & 31, w = tid >> 5;
  int half = lane >> 4, lm = lane & 15;
  int idx = blockIdx.x;
  int qb = idx & 15, head = (idx >> 4) & 7, b = idx >> 7;
  int hh0 = head * cDH;
  int mrowA = b * cS + qb * 128 + w * 16 + lm;
  Frag fQr, fQi;
#pragma unroll
  for (int vg = 0; vg < 8; vg++) {
    int dh = 2 * vg + 16 * half;
    fQr.u[vg] = *(const unsigned int*)(qr + (size_t)mrowA * cH + hh0 + dh);
    fQi.u[vg] = *(const unsigned int*)(qi + (size_t)mrowA * cH + hh0 + dh);
  }
  const float sc = 0.1767766952966369f;  // 1/sqrt(32)
  float mst[8], lst[8];
#pragma unroll
  for (int r = 0; r < 8; r++) { mst[r] = -1e30f; lst[r] = 0.f; }
  v8f o00 = {}, o01 = {}, o10 = {}, o11 = {};
  for (int t0 = 0; t0 < cS; t0 += 32) {
    __syncthreads();
    for (int s = tid; s < 512; s += 256) {
      int arr = s >> 7, slot = s & 127, row = slot >> 2, c8 = (slot & 3) * 8;
      const unsigned short* src = (arr == 0) ? kr : (arr == 1) ? ki : (arr == 2) ? vr : vi;
      unsigned short* dst = (arr == 0) ? &sK[0][row][c8] : (arr == 1) ? &sK[1][row][c8]
                          : (arr == 2) ? &sV[0][row][c8] : &sV[1][row][c8];
      async_ld16(dst, src + (size_t)(b * cS + t0 + row) * cH + hh0 + c8);
    }
    async_wait0();
    __syncthreads();
    v8f s0 = {}, s1 = {};
    {
      Frag fKr, fKi;
      const unsigned int* r0 = (const unsigned int*)&sK[0][lm][0];
      const unsigned int* r1 = (const unsigned int*)&sK[1][lm][0];
#pragma unroll
      for (int j = 0; j < 8; j++) { fKr.u[j] = r0[j + 8 * half]; fKi.u[j] = r1[j + 8 * half]; }
      s0 = bwmma(fQr, fKr, s0);
      s0 = bwmma(fQi, fKi, s0);
      const unsigned int* r2 = (const unsigned int*)&sK[0][16 + lm][0];
      const unsigned int* r3 = (const unsigned int*)&sK[1][16 + lm][0];
#pragma unroll
      for (int j = 0; j < 8; j++) { fKr.u[j] = r2[j + 8 * half]; fKi.u[j] = r3[j + 8 * half]; }
      s1 = bwmma(fQr, fKr, s1);
      s1 = bwmma(fQi, fKi, s1);
    }
    float alpha[8];
#pragma unroll
    for (int r = 0; r < 8; r++) {
      float a = s0[r] * sc, bb = s1[r] * sc;
      float mx = fmaxf(a, bb);
#pragma unroll
      for (int o = 8; o > 0; o >>= 1) mx = fmaxf(mx, __shfl_xor(mx, o, 32));
      float nm = fmaxf(mst[r], mx);
      alpha[r] = __expf(mst[r] - nm);
      mst[r] = nm;
      float p0 = __expf(a - nm), p1 = __expf(bb - nm);
      float ps = p0 + p1;
#pragma unroll
      for (int o = 8; o > 0; o >>= 1) ps += __shfl_xor(ps, o, 32);
      lst[r] = lst[r] * alpha[r] + ps;
      sP[w][r + 8 * half][lm] = f2bf(p0);
      sP[w][r + 8 * half][16 + lm] = f2bf(p1);
    }
#pragma unroll
    for (int r = 0; r < 8; r++) {
      o00[r] *= alpha[r]; o01[r] *= alpha[r]; o10[r] *= alpha[r]; o11[r] *= alpha[r];
    }
    asm volatile("" ::: "memory");  // order per-wave LDS P writes before reads
    Frag fP;
    {
      const unsigned int* rp = (const unsigned int*)&sP[w][lm][0];
#pragma unroll
      for (int vg = 0; vg < 8; vg++) {
        int p = vg + 8 * half;
        fP.u[vg] = rp[p];
      }
    }
    Frag fV;
#pragma unroll
    for (int e = 0; e < 16; e++) fV.h[e] = sV[0][16 * half + e][lm];
    o00 = bwmma(fP, fV, o00);
#pragma unroll
    for (int e = 0; e < 16; e++) fV.h[e] = sV[0][16 * half + e][16 + lm];
    o01 = bwmma(fP, fV, o01);
#pragma unroll
    for (int e = 0; e < 16; e++) fV.h[e] = sV[1][16 * half + e][lm];
    o10 = bwmma(fP, fV, o10);
#pragma unroll
    for (int e = 0; e < 16; e++) fV.h[e] = sV[1][16 * half + e][16 + lm];
    o11 = bwmma(fP, fV, o11);
  }
#pragma unroll
  for (int r = 0; r < 8; r++) {
    float inv = 1.f / lst[r];
    size_t row = (size_t)(b * cS + qb * 128 + w * 16 + r + 8 * half);
    aor[row * cH + hh0 + lm]      = f2bf(o00[r] * inv);
    aor[row * cH + hh0 + 16 + lm] = f2bf(o01[r] * inv);
    aoi[row * cH + hh0 + lm]      = f2bf(o10[r] * inv);
    aoi[row * cH + hh0 + 16 + lm] = f2bf(o11[r] * inv);
  }
}

// ---------------------------------------------------------------- final head
__global__ __launch_bounds__(256) void k_final(
    const float* __restrict__ hr, const float* __restrict__ hi,
    const float* __restrict__ fmodR, const float* __restrict__ fmodI,
    const float* __restrict__ fWr, const float* __restrict__ fWi,
    const float* __restrict__ fbr, const float* __restrict__ fbi,
    float* __restrict__ out) {
  __shared__ float sbuf[8];
  int bs = blockIdx.x, hh = threadIdx.x, b = bs / cS;
  float vr = hr[(size_t)bs * cH + hh], vi = hi[(size_t)bs * cH + hh];
  vr = (vr != vr) ? 0.f : (isinf(vr) ? (vr > 0.f ? 1000.f : -1000.f) : vr);
  vi = (vi != vi) ? 0.f : (isinf(vi) ? (vi > 0.f ? 1000.f : -1000.f) : vi);
  float mr = blockSum(vr, sbuf) * (1.f / cH);
  float mi = blockSum(vi, sbuf) * (1.f / cH);
  float qr = blockSum(vr * vr, sbuf) * (1.f / cH);
  float qi = blockSum(vi * vi, sbuf) * (1.f / cH);
  float nr = (vr - mr) * rsqrtf(qr - mr * mr + 1e-6f);
  float ni = (vi - mi) * rsqrtf(qi - mi * mi + 1e-6f);
  float shr = clampf(fmodR[(size_t)b * 2 * cH + hh], -2.f, 2.f);
  float shi = clampf(fmodI[(size_t)b * 2 * cH + hh], -2.f, 2.f);
  float scr = clampf(fmodR[(size_t)b * 2 * cH + cH + hh], -2.f, 2.f);
  float sci = clampf(fmodI[(size_t)b * 2 * cH + cH + hh], -2.f, 2.f);
  float yr = clampf(nr * (1.f + scr) + shr, -10.f, 10.f);
  float yi = clampf(ni * (1.f + sci) + shi, -10.f, 10.f);
  float wr = fWr[hh], wi = fWi[hh];
  float pr = blockSum(yr * wr - yi * wi, sbuf);
  float pi = blockSum(yi * wr + yr * wi, sbuf);
  if (hh == 0) {
    pr += fbr[0]; pi += fbi[0];
    out[(size_t)bs * 2 + 0] = tanhf(pr * 0.25f) * 4.f;
    out[(size_t)bs * 2 + 1] = tanhf(pi * 0.25f) * 4.f;
  }
}

// ------------------------------------------------------------------- driver
extern "C" void kernel_launch(void* const* d_in, const int* in_sizes, int n_in,
                              void* d_out, int out_size, void* d_ws, size_t ws_size,
                              hipStream_t stream) {
  (void)in_sizes; (void)n_in; (void)out_size; (void)ws_size;
  const float* x  = (const float*)d_in[0];
  const int*   tt = (const int*)d_in[1];
  const float* cin = (const float*)d_in[2];
  auto L = [&](int i) -> const float* { return (const float*)d_in[i]; };
  // leaf indices (insertion order): p_lin=3, t_lin1=7, t_lin2=11,
  // c_lin1=15, c_lin2=19, c_lin3=23, blocks base=27 (+28/blk:
  // ada+0,q+4,k+8,v+12,o+16,m1+20,m2+24), f_ada=139, f_lin=143
  char* wp = (char*)d_ws;
  auto alloc = [&](size_t bytes) -> char* {
    char* r = wp; wp += (bytes + 255) & ~(size_t)255; return r;
  };
  float* hr  = (float*)alloc(sizeof(float) * (size_t)cM * cH);
  float* hi  = (float*)alloc(sizeof(float) * (size_t)cM * cH);
  float* ter = (float*)alloc(sizeof(float) * cB * cEMB);
  float* tei = (float*)alloc(sizeof(float) * cB * cEMB);
  float* t1r = (float*)alloc(sizeof(float) * cB * cH);
  float* t1i = (float*)alloc(sizeof(float) * cB * cH);
  float* c1r = (float*)alloc(sizeof(float) * cB * cH);
  float* c1i = (float*)alloc(sizeof(float) * cB * cH);
  float* c2r = (float*)alloc(sizeof(float) * cB * cH4);
  float* c2i = (float*)alloc(sizeof(float) * cB * cH4);
  float* ccr = (float*)alloc(sizeof(float) * cB * cH);
  float* cci = (float*)alloc(sizeof(float) * cB * cH);
  float* modR = (float*)alloc(sizeof(float) * cB * 6 * cH);
  float* modI = (float*)alloc(sizeof(float) * cB * 6 * cH);
  float* fmR = (float*)alloc(sizeof(float) * cB * 2 * cH);
  float* fmI = (float*)alloc(sizeof(float) * cB * 2 * cH);
  unsigned short* mxr = (unsigned short*)alloc(2ull * cM * cH);
  unsigned short* mxi = (unsigned short*)alloc(2ull * cM * cH);
  unsigned short* qpr = (unsigned short*)alloc(2ull * cM * cH);
  unsigned short* qpi = (unsigned short*)alloc(2ull * cM * cH);
  unsigned short* kpr = (unsigned short*)alloc(2ull * cM * cH);
  unsigned short* kpi = (unsigned short*)alloc(2ull * cM * cH);
  unsigned short* vpr = (unsigned short*)alloc(2ull * cM * cH);
  unsigned short* vpi = (unsigned short*)alloc(2ull * cM * cH);
  unsigned short* aor = (unsigned short*)alloc(2ull * cM * cH);
  unsigned short* aoi = (unsigned short*)alloc(2ull * cM * cH);
  unsigned short* mhr = (unsigned short*)alloc(2ull * cM * cH4);
  unsigned short* mhi = (unsigned short*)alloc(2ull * cM * cH4);
  unsigned short* WB[cL][12];
  for (int blk = 0; blk < cL; blk++) {
    for (int j = 0; j < 8; j++)   WB[blk][j] = (unsigned short*)alloc(2ull * cH * cH);
    for (int j = 8; j < 10; j++)  WB[blk][j] = (unsigned short*)alloc(2ull * cH4 * cH);
    for (int j = 10; j < 12; j++) WB[blk][j] = (unsigned short*)alloc(2ull * cH * cH4);
  }
  auto cvt = [&](const float* src, unsigned short* dst, int n) {
    k_f32_to_bf16<<<(n + 255) / 256, 256, 0, stream>>>(src, dst, n);
  };
  for (int blk = 0; blk < cL; blk++) {
    int base = 27 + 28 * blk;
    cvt(L(base + 4), WB[blk][0], cH * cH);    cvt(L(base + 5), WB[blk][1], cH * cH);     // q
    cvt(L(base + 8), WB[blk][2], cH * cH);    cvt(L(base + 9), WB[blk][3], cH * cH);     // k
    cvt(L(base + 12), WB[blk][4], cH * cH);   cvt(L(base + 13), WB[blk][5], cH * cH);    // v
    cvt(L(base + 16), WB[blk][6], cH * cH);   cvt(L(base + 17), WB[blk][7], cH * cH);    // o
    cvt(L(base + 20), WB[blk][8], cH4 * cH);  cvt(L(base + 21), WB[blk][9], cH4 * cH);   // m1
    cvt(L(base + 24), WB[blk][10], cH * cH4); cvt(L(base + 25), WB[blk][11], cH * cH4);  // m2
  }
  // position embedding
  k_embed<<<cM, 256, 0, stream>>>(x, L(3), L(4), L(5), L(6), hr, hi);
  // conditioning vector cc
  k_te<<<dim3(1, cB), cEMB, 0, stream>>>(tt, ter, tei);
  k_clin_vec<<<dim3(1, cB), 256, 0, stream>>>(ter, tei, cEMB, L(7), L(8), L(9), L(10),
                                              t1r, t1i, cH, cH, cEMB, 0, 0, 0);
  k_clin_vec<<<dim3(1, cB), 256, 0, stream>>>(t1r, t1i, cH, L(11), L(12), L(13), L(14),
                                              ccr, cci, cH, cH, cH, 1, 0, 0);
  k_clin_vec<<<dim3(1, cB), 256, 0, stream>>>(cin, cin, cCOND, L(15), L(16), L(17), L(18),
                                              c1r, c1i, cH, cH, cCOND, 0, 1, 0);
  k_clin_vec<<<dim3(4, cB), 256, 0, stream>>>(c1r, c1i, cH, L(19), L(20), L(21), L(22),
                                              c2r, c2i, cH4, cH4, cH, 1, 0, 0);
  k_clin_vec<<<dim3(1, cB), 256, 0, stream>>>(c2r, c2i, cH4, L(23), L(24), L(25), L(26),
                                              ccr, cci, cH, cH, cH4, 1, 0, 1);
  // transformer blocks
  for (int blk = 0; blk < cL; blk++) {
    int base = 27 + 28 * blk;
    k_clin_vec<<<dim3(6, cB), 256, 0, stream>>>(ccr, cci, cH, L(base + 0), L(base + 1),
                                                L(base + 2), L(base + 3),
                                                modR, modI, 6 * cH, 6 * cH, cH, 1, 0, 0);
    // ---- attention half
    k_ln_mod<<<cM, 256, 0, stream>>>(hr, hi, modR, modI, 0, 1, mxr, mxi);
    dim3 gH(cH / 64, cM / 128);     // 4 x 32
    dim3 gM1(cH4 / 64, cM / 128);   // 16 x 32
    k_cgemm<0, 0><<<gH, 256, 0, stream>>>(mxr, mxi, WB[blk][0], WB[blk][1],
                                          L(base + 6), L(base + 7),
                                          qpr, qpi, nullptr, nullptr, nullptr, nullptr, 0, cH, cH);
    k_cgemm<0, 0><<<gH, 256, 0, stream>>>(mxr, mxi, WB[blk][2], WB[blk][3],
                                          L(base + 10), L(base + 11),
                                          kpr, kpi, nullptr, nullptr, nullptr, nullptr, 0, cH, cH);
    k_cgemm<0, 0><<<gH, 256, 0, stream>>>(mxr, mxi, WB[blk][4], WB[blk][5],
                                          L(base + 14), L(base + 15),
                                          vpr, vpi, nullptr, nullptr, nullptr, nullptr, 0, cH, cH);
    k_attn<<<cB * cNH * (cS / 128), 256, 0, stream>>>(qpr, qpi, kpr, kpi, vpr, vpi, aor, aoi);
    k_cgemm<1, 0><<<gH, 256, 0, stream>>>(aor, aoi, WB[blk][6], WB[blk][7],
                                          L(base + 18), L(base + 19),
                                          nullptr, nullptr, hr, hi,
                                          modR + 2 * cH, modI + 2 * cH, 6 * cH, cH, cH);
    // ---- MLP half
    k_ln_mod<<<cM, 256, 0, stream>>>(hr, hi, modR, modI, 3, 4, mxr, mxi);
    k_cgemm<0, 1><<<gM1, 256, 0, stream>>>(mxr, mxi, WB[blk][8], WB[blk][9],
                                           L(base + 22), L(base + 23),
                                           mhr, mhi, nullptr, nullptr, nullptr, nullptr, 0, cH4, cH);
    k_cgemm<1, 0><<<gH, 256, 0, stream>>>(mhr, mhi, WB[blk][10], WB[blk][11],
                                          L(base + 26), L(base + 27),
                                          nullptr, nullptr, hr, hi,
                                          modR + 5 * cH, modI + 5 * cH, 6 * cH, cH, cH4);
  }
  // final layer
  k_clin_vec<<<dim3(2, cB), 256, 0, stream>>>(ccr, cci, cH, L(139), L(140), L(141), L(142),
                                              fmR, fmI, 2 * cH, 2 * cH, cH, 1, 0, 0);
  k_final<<<cM, 256, 0, stream>>>(hr, hi, fmR, fmI, L(143), L(144), L(145), L(146),
                                  (float*)d_out);
}
